// CoPELoss_80891414052987
// MI455X (gfx1250) — compile-verified, matrix operand order checked by reading, not run on hardware
//
#include <hip/hip_runtime.h>
#include <hip/hip_bf16.h>
#include <math.h>

typedef unsigned short u16;
typedef unsigned int   u32;
typedef __attribute__((ext_vector_type(16))) __bf16 v16bf;
typedef __attribute__((ext_vector_type(8)))  float  v8f;

#define N_ROWS 8192
#define M_ROWS 8192
#define DDIM   128
#define KD     256      // packed feature dim: [mu | sigma]
#define TILE   128
#define LDSS   264      // LDS row stride in bf16 elems (256 + 8 pad -> conflict-free)

__device__ __forceinline__ u16 f2bf(float f) {
  // round-to-nearest-even float -> bf16
  u32 u = __float_as_uint(f);
  u32 r = u + 0x7FFFu + ((u >> 16) & 1u);
  return (u16)(r >> 16);
}

// Pack [mu | sigma] rows as bf16; emit row sq-norm over both halves and mean(sigma).
__global__ __launch_bounds__(256) void prep_kernel(
    const float* __restrict__ mu, const float* __restrict__ var,
    u16* __restrict__ packed, float* __restrict__ rnorm, float* __restrict__ sbar,
    int rows) {
  const int row  = blockIdx.x * 8 + (threadIdx.x >> 5);
  const int lane = threadIdx.x & 31;
  if (row >= rows) return;

  const float4* mu4 = (const float4*)(mu  + (size_t)row * DDIM);
  const float4* va4 = (const float4*)(var + (size_t)row * DDIM);
  float4 m = mu4[lane];
  float4 v = va4[lane];
  float4 s;
  s.x = expf(0.5f * v.x);
  s.y = expf(0.5f * v.y);
  s.z = expf(0.5f * v.z);
  s.w = expf(0.5f * v.w);

  float nrm  = m.x*m.x + m.y*m.y + m.z*m.z + m.w*m.w
             + s.x*s.x + s.y*s.y + s.z*s.z + s.w*s.w;
  float ssum = s.x + s.y + s.z + s.w;
#pragma unroll
  for (int off = 16; off >= 1; off >>= 1) {
    nrm  += __shfl_xor(nrm,  off, 32);
    ssum += __shfl_xor(ssum, off, 32);
  }

  uint2 pm, ps;
  pm.x = (u32)f2bf(m.x) | ((u32)f2bf(m.y) << 16);
  pm.y = (u32)f2bf(m.z) | ((u32)f2bf(m.w) << 16);
  ps.x = (u32)f2bf(s.x) | ((u32)f2bf(s.y) << 16);
  ps.y = (u32)f2bf(s.z) | ((u32)f2bf(s.w) << 16);
  *(uint2*)(packed + (size_t)row * KD + lane * 4)        = pm;
  *(uint2*)(packed + (size_t)row * KD + DDIM + lane * 4) = ps;

  if (lane == 0) {
    rnorm[row] = nrm;
    sbar[row]  = ssum * (1.0f / DDIM);
  }
}

// Load one 16x32 bf16 WMMA operand fragment from LDS (ISA 16-bit A layout):
// lanes 0-15: K k0..k0+7 then k0+16..k0+23 ; lanes 16-31 pass k0+8 in.
__device__ __forceinline__ v16bf frag_ld(const u16* p) {
  union { uint4 q[2]; v16bf v; } u;
  u.q[0] = *(const uint4*)(p);
  u.q[1] = *(const uint4*)(p + 16);
  return u.v;
}

#define WMMA_BF16(A_, B_, C_) \
  __builtin_amdgcn_wmma_f32_16x16x32_bf16(false, (A_), false, (B_), (short)0, (C_), false, false)

__global__ __launch_bounds__(256) void gemm_bce_kernel(
    const u16* __restrict__ A, const u16* __restrict__ B,
    const float* __restrict__ r1, const float* __restrict__ r2,
    const float* __restrict__ s1, const float* __restrict__ s2,
    const float* __restrict__ matched,
    const float* __restrict__ shiftp, const float* __restrict__ scalep,
    double* __restrict__ partials) {
  __shared__ u16 lA[TILE * LDSS];
  __shared__ u16 lB[TILE * LDSS];
  __shared__ double red[8];

  const int t  = threadIdx.x;
  const int bx = blockIdx.x;   // column tile over M
  const int by = blockIdx.y;   // row tile over N

  // ---- stage full-K tiles into LDS (A: 128x256 bf16, B: 128x256 bf16) ----
  const u16* gA = A + (size_t)by * TILE * KD;
  const u16* gB = B + (size_t)bx * TILE * KD;
#pragma unroll
  for (int i = 0; i < 16; ++i) {
    const int c   = t + 256 * i;       // 16B-chunk id, 4096 chunks per tile
    const int row = c >> 5;            // 32 chunks per row
    const int col = (c & 31) * 8;      // bf16 element offset
    *(uint4*)(&lA[row * LDSS + col]) = *(const uint4*)(gA + row * KD + col);
    *(uint4*)(&lB[row * LDSS + col]) = *(const uint4*)(gB + row * KD + col);
  }

  // ---- prefetch this block's 64KB `matched` tile while WMMAs run ----
  // 128 rows x 512B; 4 x 128B cachelines per row -> 512 prefetches, 2/thread.
  {
    const float* mbase = matched + (size_t)(by * TILE) * M_ROWS + bx * TILE;
#pragma unroll
    for (int p = 0; p < 2; ++p) {
      const int idx = t + p * 256;       // 0..511
      const int r   = idx >> 2;          // row 0..127
      const int seg = (idx & 3) * 32;    // float offset 0,32,64,96
      __builtin_prefetch(mbase + (size_t)r * M_ROWS + seg, 0, 1);
    }
  }
  __syncthreads();

  const int lane  = t & 31;
  const int wave  = t >> 5;
  const int wm    = wave & 3;          // 2 subtile rows per wave
  const int wn    = wave >> 2;         // 4 subtile cols per wave
  const int mrow  = lane & 15;
  const int halfK = (lane >> 4) * 8;

  const v8f vzero = {0.f, 0.f, 0.f, 0.f, 0.f, 0.f, 0.f, 0.f};
  v8f acc[2][4];
#pragma unroll
  for (int i = 0; i < 2; ++i)
#pragma unroll
    for (int j = 0; j < 4; ++j) acc[i][j] = vzero;

  const u16* rowA0 = &lA[((wm * 2 + 0) * 16 + mrow) * LDSS + halfK];
  const u16* rowA1 = &lA[((wm * 2 + 1) * 16 + mrow) * LDSS + halfK];
  const u16* rowB0 = &lB[((wn * 4 + 0) * 16 + mrow) * LDSS + halfK];
  const u16* rowB1 = &lB[((wn * 4 + 1) * 16 + mrow) * LDSS + halfK];
  const u16* rowB2 = &lB[((wn * 4 + 2) * 16 + mrow) * LDSS + halfK];
  const u16* rowB3 = &lB[((wn * 4 + 3) * 16 + mrow) * LDSS + halfK];

#pragma unroll
  for (int kb = 0; kb < 8; ++kb) {
    const int k0 = kb * 32;
    v16bf a0 = frag_ld(rowA0 + k0);
    v16bf a1 = frag_ld(rowA1 + k0);
    v16bf b0 = frag_ld(rowB0 + k0);
    v16bf b1 = frag_ld(rowB1 + k0);
    v16bf b2 = frag_ld(rowB2 + k0);
    v16bf b3 = frag_ld(rowB3 + k0);
    acc[0][0] = WMMA_BF16(a0, b0, acc[0][0]);
    acc[0][1] = WMMA_BF16(a0, b1, acc[0][1]);
    acc[0][2] = WMMA_BF16(a0, b2, acc[0][2]);
    acc[0][3] = WMMA_BF16(a0, b3, acc[0][3]);
    acc[1][0] = WMMA_BF16(a1, b0, acc[1][0]);
    acc[1][1] = WMMA_BF16(a1, b1, acc[1][1]);
    acc[1][2] = WMMA_BF16(a1, b2, acc[1][2]);
    acc[1][3] = WMMA_BF16(a1, b3, acc[1][3]);
  }

  // ---- fused epilogue: distance -> logits -> BCE-with-logits partial sum ----
  // Uses hardware transcendentals (v_exp_f32/v_log_f32). log(1+e) with e in
  // (0,1]: no cancellation; underflow of e only when |l|>16 where the true
  // term is < 1.2e-7 -> negligible in a 67M-element mean.
  const float shiftv = shiftp[0];
  const float scalev = scalep[0];
  double lsum = 0.0;
#pragma unroll
  for (int i = 0; i < 2; ++i) {
    // C/D layout: lane L -> N = L%16 ; VGPR v -> M = v + 8*(L/16)
    const int gi0 = by * TILE + (wm * 2 + i) * 16 + (lane >> 4) * 8;
    float r1v[8], s1v[8];
#pragma unroll
    for (int v = 0; v < 8; ++v) { r1v[v] = r1[gi0 + v]; s1v[v] = s1[gi0 + v]; }
#pragma unroll
    for (int j = 0; j < 4; ++j) {
      const int gj  = bx * TILE + (wn * 4 + j) * 16 + (lane & 15);
      const float r2j = r2[gj];
      const float s2j = s2[gj];
      const float* mcol = matched + gj;
      float fs = 0.0f;
#pragma unroll
      for (int v = 0; v < 8; ++v) {
        const float ab   = acc[i][j][v];
        // term1+term2 = r1 + r2 - 2*ab ; term3 = 2*D*s1bar*s2bar (2*128 = 256)
        const float dist  = r1v[v] + r2j + 256.0f * s1v[v] * s2j - 2.0f * ab;
        const float logit = fmaf(-scalev, dist, shiftv);
        const float tgt   = mcol[(size_t)(gi0 + v) * M_ROWS];
        const float bce   = fmaxf(logit, 0.0f) - logit * tgt
                          + __logf(1.0f + __expf(-fabsf(logit)));
        fs += bce;
      }
      lsum += (double)fs;
    }
  }

  // deterministic block reduction: wave shuffle tree, then wave 0 sums 8 values
#pragma unroll
  for (int off = 16; off >= 1; off >>= 1) lsum += __shfl_xor(lsum, off, 32);
  if (lane == 0) red[wave] = lsum;
  __syncthreads();
  if (t == 0) {
    double tot = 0.0;
#pragma unroll
    for (int w = 0; w < 8; ++w) tot += red[w];
    partials[by * gridDim.x + bx] = tot;
  }
}

__global__ __launch_bounds__(256) void final_reduce_kernel(
    const double* __restrict__ partials, int n, float* __restrict__ out) {
  __shared__ double red[256];
  double s = 0.0;
  for (int i = threadIdx.x; i < n; i += 256) s += partials[i];
  red[threadIdx.x] = s;
  __syncthreads();
  for (int off = 128; off > 0; off >>= 1) {
    if (threadIdx.x < off) red[threadIdx.x] += red[threadIdx.x + off];
    __syncthreads();
  }
  if (threadIdx.x == 0)
    out[0] = (float)(red[0] * (1.0 / ((double)N_ROWS * (double)M_ROWS)));
}

extern "C" void kernel_launch(void* const* d_in, const int* in_sizes, int n_in,
                              void* d_out, int out_size, void* d_ws, size_t ws_size,
                              hipStream_t stream) {
  const float* mu1      = (const float*)d_in[0];
  const float* var1     = (const float*)d_in[1];
  const float* mu2      = (const float*)d_in[2];
  const float* var2     = (const float*)d_in[3];
  const float* matched  = (const float*)d_in[4];
  const float* shift    = (const float*)d_in[5];
  const float* negscale = (const float*)d_in[6];

  char* ws = (char*)d_ws;
  size_t off = 0;
  u16* A = (u16*)(ws + off);       off += (size_t)N_ROWS * KD * sizeof(u16);   // 4 MiB
  u16* B = (u16*)(ws + off);       off += (size_t)M_ROWS * KD * sizeof(u16);   // 4 MiB
  float* r1 = (float*)(ws + off);  off += (size_t)N_ROWS * sizeof(float);
  float* r2 = (float*)(ws + off);  off += (size_t)M_ROWS * sizeof(float);
  float* s1 = (float*)(ws + off);  off += (size_t)N_ROWS * sizeof(float);
  float* s2 = (float*)(ws + off);  off += (size_t)M_ROWS * sizeof(float);
  const int nTiles = (N_ROWS / TILE) * (M_ROWS / TILE);                        // 4096
  double* partials = (double*)(ws + off); off += (size_t)nTiles * sizeof(double);

  prep_kernel<<<N_ROWS / 8, 256, 0, stream>>>(mu1, var1, A, r1, s1, N_ROWS);
  prep_kernel<<<M_ROWS / 8, 256, 0, stream>>>(mu2, var2, B, r2, s2, M_ROWS);

  dim3 grid(M_ROWS / TILE, N_ROWS / TILE);
  gemm_bce_kernel<<<grid, 256, 0, stream>>>(A, B, r1, r2, s1, s2, matched,
                                            shift, negscale, partials);

  final_reduce_kernel<<<1, 256, 0, stream>>>(partials, nTiles, (float*)d_out);
}
